// hashedLayer_15513421873631
// MI455X (gfx1250) — compile-verified, matrix operand order checked by reading, not run on hardware
//
#include <hip/hip_runtime.h>

// Problem constants (from reference): B=4096, FAN_IN=4096, FAN_OUT=4096, K=65536
#define B_DIM 4096
#define FIN   4096
#define FOUT  4096
#define KP1   4097   // FAN_IN + 1 (bias column)
#define LDK   4128   // KP1 padded up to a multiple of 32 (WMMA K step)
#define NCG   (LDK / 8)  // 516 groups of 8 columns per row

typedef __attribute__((ext_vector_type(16))) _Float16      v16h;
typedef __attribute__((ext_vector_type(8)))  float         v8f;
typedef __attribute__((ext_vector_type(8)))  unsigned short u16x8;

static __device__ __forceinline__ unsigned short f2h_raw(float f) {
  _Float16 h = (_Float16)f;
  return __builtin_bit_cast(unsigned short, h);
}

static __device__ __forceinline__ v16h combine16(u16x8 lo, u16x8 hi) {
  union U { struct { u16x8 lo, hi; } s; v16h v; } u;
  u.s.lo = lo;
  u.s.hi = hi;
  return u.v;
}

// ---------------------------------------------------------------------------
// Kernel 1: pack A (fp32 -> f16), append bias-one column, zero-pad to LDK.
// One thread handles 8 contiguous columns -> one aligned 16B store.
// ---------------------------------------------------------------------------
__global__ __launch_bounds__(256) void hn_pack_a(const float* __restrict__ a,
                                                 unsigned short* __restrict__ Ah) {
  int t = blockIdx.x * 256 + threadIdx.x;
  if (t >= B_DIM * NCG) return;
  int row = t / NCG;
  int j0  = (t - row * NCG) * 8;
  const float* ar = a + (size_t)row * FIN;
  u16x8 v;
  if (j0 + 8 <= FIN) {
#pragma unroll
    for (int k = 0; k < 8; ++k) v[k] = f2h_raw(ar[j0 + k]);
  } else {
#pragma unroll
    for (int k = 0; k < 8; ++k) {
      int j = j0 + k;
      float f = (j < FIN) ? ar[j] : (j == FIN ? 1.0f : 0.0f);
      v[k] = f2h_raw(f);
    }
  }
  *(u16x8*)(Ah + (size_t)row * LDK + j0) = v;
}

// ---------------------------------------------------------------------------
// Kernel 2: materialize virtual weights Wv[i,j] = W[hash_idx[i,j]] as f16,
// zero-padded to LDK. W is 256KB -> L2-resident gather, done exactly once.
// ---------------------------------------------------------------------------
__global__ __launch_bounds__(256) void hn_gather_w(const int* __restrict__ hidx,
                                                   const float* __restrict__ W,
                                                   unsigned short* __restrict__ Wh) {
  int t = blockIdx.x * 256 + threadIdx.x;
  if (t >= FOUT * NCG) return;
  int row = t / NCG;
  int j0  = (t - row * NCG) * 8;
  const int* hr = hidx + (size_t)row * KP1;
  u16x8 v;
  if (j0 + 8 <= KP1) {
#pragma unroll
    for (int k = 0; k < 8; ++k) v[k] = f2h_raw(W[hr[j0 + k]]);
  } else {
#pragma unroll
    for (int k = 0; k < 8; ++k) {
      int j = j0 + k;
      v[k] = (j < KP1) ? f2h_raw(W[hr[j]]) : (unsigned short)0;
    }
  }
  *(u16x8*)(Wh + (size_t)row * LDK + j0) = v;
}

// ---------------------------------------------------------------------------
// Kernel 3: C[b,i] = sum_k Ah[b,k] * Wh[i,k]  via v_wmma_f32_16x16x32_f16.
//
// Wave32 tile plan: each wave computes a 64x64 C block as 4x4 WMMA tiles
// (128 accumulator VGPRs). 8 waves/block -> 128(M) x 256(N) block tile.
// All operands live in L2 (A_f16 + Wv_f16 + C = 131MB < 192MB); per-K-step
// block working set (128+256 rows x 32K x 2B = 24KB) fits WGP$, so cache
// captures the cross-wave operand reuse without LDS staging. Register
// blocking gives 16 WMMA per 16 x 16B loads per K-step.
//
// 16-bit operand lane layout (ISA 7.12.2): lane = 16*g + l (g=0,1; l=0..15),
// row = l, vector element e -> K = (e%8) + 16*(e/8) + 8*g, i.e. two
// contiguous 16B chunks per lane at K-offsets {8g, 16+8g}.
// C/D layout: VGPR v holds row (v + 8*g), col l.
// ---------------------------------------------------------------------------
__global__ __launch_bounds__(256) void hn_gemm_f16_wmma(
    const unsigned short* __restrict__ Ah,
    const unsigned short* __restrict__ Wh,
    float* __restrict__ out) {
  const int lane = threadIdx.x & 31;
  const int wid  = threadIdx.x >> 5;
  const int l    = lane & 15;
  const int g    = lane >> 4;

  const int m_base = blockIdx.y * 128 + (wid >> 2) * 64;  // rows of A / C
  const int n_base = blockIdx.x * 256 + (wid & 3) * 64;   // rows of Wv = cols of C

  const unsigned short* arow[4];
  const unsigned short* brow[4];
#pragma unroll
  for (int mi = 0; mi < 4; ++mi)
    arow[mi] = Ah + (size_t)(m_base + mi * 16 + l) * LDK + 8 * g;
#pragma unroll
  for (int ni = 0; ni < 4; ++ni)
    brow[ni] = Wh + (size_t)(n_base + ni * 16 + l) * LDK + 8 * g;

  v8f acc[4][4] = {};

  for (int k0 = 0; k0 < LDK; k0 += 32) {
    // WGP-scope prefetch (locality 3) of a future K-step: pulls into all
    // cache levels (incl. WGP$) on miss, warming the load clauses below.
    // OOB prefetch addresses are speculative and silently dropped.
    __builtin_prefetch(arow[0] + k0 + 512, 0, 3);
    __builtin_prefetch(brow[0] + k0 + 512, 0, 3);

    v16h af[4], bf[4];
#pragma unroll
    for (int mi = 0; mi < 4; ++mi) {
      u16x8 lo = *(const u16x8*)(arow[mi] + k0);
      u16x8 hi = *(const u16x8*)(arow[mi] + k0 + 16);
      af[mi] = combine16(lo, hi);
    }
#pragma unroll
    for (int ni = 0; ni < 4; ++ni) {
      u16x8 lo = *(const u16x8*)(brow[ni] + k0);
      u16x8 hi = *(const u16x8*)(brow[ni] + k0 + 16);
      bf[ni] = combine16(lo, hi);
    }
#pragma unroll
    for (int mi = 0; mi < 4; ++mi)
#pragma unroll
      for (int ni = 0; ni < 4; ++ni)
        acc[mi][ni] = __builtin_amdgcn_wmma_f32_16x16x32_f16(
            false, af[mi], false, bf[ni], (short)0, acc[mi][ni], false, false);
  }

  // Store: acc VGPR v holds C[m_base + mi*16 + v + 8g][n_base + ni*16 + l].
  // Lanes 0-15 / 16-31 each cover 16 contiguous floats -> 64B segments.
#pragma unroll
  for (int mi = 0; mi < 4; ++mi) {
#pragma unroll
    for (int v = 0; v < 8; ++v) {
      int row = m_base + mi * 16 + 8 * g + v;
      float* c = out + (size_t)row * FOUT + n_base + l;
#pragma unroll
      for (int ni = 0; ni < 4; ++ni)
        c[ni * 16] = acc[mi][ni][v];
    }
  }
}

// ---------------------------------------------------------------------------
extern "C" void kernel_launch(void* const* d_in, const int* in_sizes, int n_in,
                              void* d_out, int out_size, void* d_ws, size_t ws_size,
                              hipStream_t stream) {
  (void)in_sizes; (void)n_in; (void)out_size; (void)ws_size;

  const float* a    = (const float*)d_in[0];   // [4096, 4096] fp32
  const int*   hidx = (const int*)d_in[1];     // [4096, 4097] int32
  const float* W    = (const float*)d_in[2];   // [65536] fp32
  float*       out  = (float*)d_out;           // [4096, 4096] fp32

  // Workspace: A_f16 [4096][4128] then Wv_f16 [4096][4128]  (~67.6 MB total)
  unsigned short* Ah = (unsigned short*)d_ws;
  unsigned short* Wh = Ah + (size_t)B_DIM * LDK;

  const int nPack      = B_DIM * NCG;              // 2,113,536 threads
  const int packBlocks = (nPack + 255) / 256;

  hn_pack_a<<<packBlocks, 256, 0, stream>>>(a, Ah);
  hn_gather_w<<<packBlocks, 256, 0, stream>>>(hidx, W, Wh);

  dim3 grid(FOUT / 256, B_DIM / 128);              // 16 x 32 blocks, 256 thr
  hn_gemm_f16_wmma<<<grid, 256, 0, stream>>>(Ah, Wh, out);
}
